// SHRED_53214644797670
// MI455X (gfx1250) — compile-verified, matrix operand order checked by reading, not run on hardware
//
#include <hip/hip_runtime.h>
#include <hip/hip_bf16.h>

// Problem constants (from reference)
#define BB   256
#define TT   500
#define INW  32
#define HH   64
#define NG   256   // 4*H gate width
#define DD1  350
#define DD2  400
#define OUTW 50000

typedef _Float16     v16h __attribute__((ext_vector_type(16)));
typedef float        v8f  __attribute__((ext_vector_type(8)));
typedef float        v2f  __attribute__((ext_vector_type(2)));
typedef unsigned int v4u  __attribute__((ext_vector_type(4)));
typedef int          v8i  __attribute__((ext_vector_type(8)));
typedef int          v4i  __attribute__((ext_vector_type(4)));

// Branchless activations on native v_exp_f32 / v_rcp_f32 (no libm branches).
__device__ __forceinline__ float sigm(float x) {
    return __builtin_amdgcn_rcpf(1.0f + __expf(-x));
}
__device__ __forceinline__ float ftanh(float x) {
    return 2.0f * __builtin_amdgcn_rcpf(1.0f + __expf(-2.0f * x)) - 1.0f;
}

// f16 A/B fragment element -> K index mapping (16-bit A 16x32 layout, ISA 7.12.2)
__device__ __forceinline__ int kmap(int e, int hi) {
    int v = e >> 1, h = e & 1;
    int k = (v < 4) ? (v * 2 + h) : ((v - 4) * 2 + h + 16);
    return k + hi * 8;
}

// ---------------------------------------------------------------------------
// Kernel 1: two-layer LSTM scan. 16 blocks x 256 threads (8 wave32 waves).
// Block b owns batch rows [16b, 16b+16). Gate GEMM (16 x 256) per layer per
// step via v_wmma_f32_16x16x32_f16; weights pre-converted to f16 B-fragments
// held in registers for the whole 500-step scan.
// ---------------------------------------------------------------------------
__global__ __launch_bounds__(256) void lstm_scan_kernel(
    const float* __restrict__ x,
    const float* __restrict__ wih0, const float* __restrict__ whh0, const float* __restrict__ b0,
    const float* __restrict__ wih1, const float* __restrict__ whh1, const float* __restrict__ b1,
    float* __restrict__ h1_out)
{
    __shared__ float    s_lin[16][NG];   // 16 KB   gate pre-activations
    __shared__ float    s_c0[16][HH];    // 4 KB
    __shared__ float    s_c1[16][HH];    // 4 KB
    __shared__ float    s_h1f[16][HH];   // 4 KB    f32 copy of h1 (final output)
    __shared__ _Float16 s_h0h[16][HH];   // 2 KB
    __shared__ _Float16 s_h1h[16][HH];   // 2 KB
    __shared__ _Float16 s_xh[16][INW];   // 1 KB

    const int tid  = threadIdx.x;
    const int lane = tid & 31;
    const int wv   = tid >> 5;           // wave id 0..7
    const int lcol = lane & 15;
    const int hi   = lane >> 4;          // 0 for lanes 0-15, 1 for 16-31
    const int mrow0 = blockIdx.x * 16;

    // zero-init states
    #pragma unroll
    for (int j = 0; j < 4; ++j) {
        int cell = tid + 256 * j;        // 1024 cells
        int mm = cell >> 6, u = cell & 63;
        s_c0[mm][u] = 0.0f; s_c1[mm][u] = 0.0f; s_h1f[mm][u] = 0.0f;
        s_h0h[mm][u] = (_Float16)0.0f; s_h1h[mm][u] = (_Float16)0.0f;
    }

    // ---- preload weight B-fragments (K x N tiles of W^T) into registers ----
    v16h Bih0[2];            // layer0 input weights,  K=32
    v16h Bhh0[2][2];         // layer0 hidden weights, K=64 (2 frags)
    v16h Bih1[2][2];         // layer1 input (=h0) weights, K=64
    v16h Bhh1[2][2];         // layer1 hidden weights, K=64
    float bias0[2], bias1[2];

    #pragma unroll
    for (int t2 = 0; t2 < 2; ++t2) {
        int ng = (2 * wv + t2) * 16 + lcol;   // global gate column
        bias0[t2] = b0[ng];
        bias1[t2] = b1[ng];
        #pragma unroll
        for (int e = 0; e < 16; ++e) {
            int k = kmap(e, hi);
            Bih0[t2][e]    = (_Float16)wih0[ng * INW + k];
            Bhh0[t2][0][e] = (_Float16)whh0[ng * HH + k];
            Bhh0[t2][1][e] = (_Float16)whh0[ng * HH + k + 32];
            Bih1[t2][0][e] = (_Float16)wih1[ng * HH + k];
            Bih1[t2][1][e] = (_Float16)wih1[ng * HH + k + 32];
            Bhh1[t2][0][e] = (_Float16)whh1[ng * HH + k];
            Bhh1[t2][1][e] = (_Float16)whh1[ng * HH + k + 32];
        }
    }
    __syncthreads();

    for (int t = 0; t < TT; ++t) {
        // stage x_t (16 x 32) as f16
        #pragma unroll
        for (int j = 0; j < 2; ++j) {
            int p = tid + 256 * j;               // 512 elements
            int mm = p >> 5, kk = p & 31;
            s_xh[mm][kk] = (_Float16)x[((size_t)(mrow0 + mm) * TT + t) * INW + kk];
        }
        __syncthreads();

        // ---------------- layer 0 gates ----------------
        v16h Ax, Ah0a, Ah0b;
        #pragma unroll
        for (int e = 0; e < 16; ++e) {
            int k = kmap(e, hi);
            Ax[e]   = s_xh[lcol][k];
            Ah0a[e] = s_h0h[lcol][k];
            Ah0b[e] = s_h0h[lcol][k + 32];
        }
        #pragma unroll
        for (int t2 = 0; t2 < 2; ++t2) {
            v8f acc;
            #pragma unroll
            for (int r = 0; r < 8; ++r) acc[r] = bias0[t2];
            acc = __builtin_amdgcn_wmma_f32_16x16x32_f16(false, Ax,   false, Bih0[t2],    (short)0, acc, false, false);
            acc = __builtin_amdgcn_wmma_f32_16x16x32_f16(false, Ah0a, false, Bhh0[t2][0], (short)0, acc, false, false);
            acc = __builtin_amdgcn_wmma_f32_16x16x32_f16(false, Ah0b, false, Bhh0[t2][1], (short)0, acc, false, false);
            #pragma unroll
            for (int r = 0; r < 8; ++r)
                s_lin[r + hi * 8][(2 * wv + t2) * 16 + lcol] = acc[r];
        }
        __syncthreads();

        // elementwise cell update, layer 0
        #pragma unroll
        for (int j = 0; j < 4; ++j) {
            int cell = tid + 256 * j;
            int mm = cell >> 6, u = cell & 63;
            float iv = s_lin[mm][u], fv = s_lin[mm][64 + u];
            float gv = s_lin[mm][128 + u], ov = s_lin[mm][192 + u];
            float c  = s_c0[mm][u];
            float cn = sigm(fv) * c + sigm(iv) * ftanh(gv);
            float hn = sigm(ov) * ftanh(cn);
            s_c0[mm][u]  = cn;
            s_h0h[mm][u] = (_Float16)hn;
        }
        __syncthreads();

        // ---------------- layer 1 gates ----------------
        v16h A1a, A1b, Ah1a, Ah1b;
        #pragma unroll
        for (int e = 0; e < 16; ++e) {
            int k = kmap(e, hi);
            A1a[e]  = s_h0h[lcol][k];
            A1b[e]  = s_h0h[lcol][k + 32];
            Ah1a[e] = s_h1h[lcol][k];
            Ah1b[e] = s_h1h[lcol][k + 32];
        }
        #pragma unroll
        for (int t2 = 0; t2 < 2; ++t2) {
            v8f acc;
            #pragma unroll
            for (int r = 0; r < 8; ++r) acc[r] = bias1[t2];
            acc = __builtin_amdgcn_wmma_f32_16x16x32_f16(false, A1a,  false, Bih1[t2][0], (short)0, acc, false, false);
            acc = __builtin_amdgcn_wmma_f32_16x16x32_f16(false, A1b,  false, Bih1[t2][1], (short)0, acc, false, false);
            acc = __builtin_amdgcn_wmma_f32_16x16x32_f16(false, Ah1a, false, Bhh1[t2][0], (short)0, acc, false, false);
            acc = __builtin_amdgcn_wmma_f32_16x16x32_f16(false, Ah1b, false, Bhh1[t2][1], (short)0, acc, false, false);
            #pragma unroll
            for (int r = 0; r < 8; ++r)
                s_lin[r + hi * 8][(2 * wv + t2) * 16 + lcol] = acc[r];
        }
        __syncthreads();

        // elementwise cell update, layer 1
        #pragma unroll
        for (int j = 0; j < 4; ++j) {
            int cell = tid + 256 * j;
            int mm = cell >> 6, u = cell & 63;
            float iv = s_lin[mm][u], fv = s_lin[mm][64 + u];
            float gv = s_lin[mm][128 + u], ov = s_lin[mm][192 + u];
            float c  = s_c1[mm][u];
            float cn = sigm(fv) * c + sigm(iv) * ftanh(gv);
            float hn = sigm(ov) * ftanh(cn);
            s_c1[mm][u]  = cn;
            s_h1h[mm][u] = (_Float16)hn;
            s_h1f[mm][u] = hn;
        }
        __syncthreads();
    }

    // write final h1 (f32)
    #pragma unroll
    for (int j = 0; j < 4; ++j) {
        int cell = tid + 256 * j;
        int mm = cell >> 6, u = cell & 63;
        h1_out[(size_t)(mrow0 + mm) * HH + u] = s_h1f[mm][u];
    }
}

// ---------------------------------------------------------------------------
// Decoder layers 0/1: small (<= 36 MFLOP), plain VALU, one thread per output.
// ---------------------------------------------------------------------------
__global__ __launch_bounds__(256) void dec0_kernel(
    const float* __restrict__ h1, const float* __restrict__ dw0,
    const float* __restrict__ db0, float* __restrict__ a0)
{
    int idx = blockIdx.x * 256 + threadIdx.x;
    if (idx >= BB * DD1) return;
    int m = idx / DD1, n = idx % DD1;
    float s = db0[n];
    #pragma unroll 8
    for (int k = 0; k < HH; ++k) s += h1[m * HH + k] * dw0[n * HH + k];
    a0[idx] = fmaxf(s, 0.0f);
}

__global__ __launch_bounds__(256) void dec1_kernel(
    const float* __restrict__ a0, const float* __restrict__ dw1,
    const float* __restrict__ db1, float* __restrict__ a1)
{
    int idx = blockIdx.x * 256 + threadIdx.x;
    if (idx >= BB * DD2) return;
    int m = idx / DD2, n = idx % DD2;
    float s = db1[n];
    #pragma unroll 5
    for (int k = 0; k < DD1; ++k) s += a0[m * DD1 + k] * dw1[n * DD1 + k];
    a1[idx] = fmaxf(s, 0.0f);
}

// ---------------------------------------------------------------------------
// Decoder layer 2: (256 x 400) x (400 x 50000) via v_wmma_f32_16x16x4_f32.
// 3125 blocks. The 16x400 fp32 strip of dw2 is DMA'd into LDS by the Tensor
// Data Mover (one tensor_load_to_lds per block, wave 0 only), tracked with
// TENSORcnt. K=400 = 100 x 4 maps exactly onto the f32 WMMA.
// ---------------------------------------------------------------------------
__global__ __launch_bounds__(256) void dec2_kernel(
    const float* __restrict__ a1, const float* __restrict__ dw2,
    const float* __restrict__ db2, float* __restrict__ out)
{
    __shared__ float sB[16][DD2];   // 25.6 KB: dw2 rows [16*nt, 16*nt+16)

    const int nt   = blockIdx.x;          // N tile, 0..3124
    const int tid  = threadIdx.x;
    const int lane = tid & 31;
    const int wv   = tid >> 5;
    const int lcol = lane & 15;
    const int hi   = lane >> 4;
    const int koff = hi * 2;              // f32 A/B layout: lanes 16-31 hold K+2,K+3

    if (wv == 0) {
        // --- Tensor DMA descriptor (ISA cdna5 ch.8): 2D tile 400x16 of the
        // 400x50000 (dim0 x dim1) fp32 tensor dw2, row stride 400. ---
        unsigned lds_off = (unsigned)(size_t)&sB[0][0];        // low 32 bits = LDS offset
        unsigned long long ga =
            (unsigned long long)(size_t)(dw2 + (size_t)nt * 16 * DD2);
        v4u g0;
        g0[0] = 1u;                                            // count=1 (valid), user mode
        g0[1] = lds_off;                                       // lds_addr
        g0[2] = (unsigned)(ga & 0xFFFFFFFFu);                  // global_addr[31:0]
        g0[3] = (unsigned)((ga >> 32) & 0x01FFFFFFu) | 0x80000000u; // addr[56:32] | type=2
        v8i g1;
        g1[0] = (int)0x00020000;                               // data_size=2 (4 bytes)
        g1[1] = (int)((unsigned)DD2 << 16);                    // tensor_dim0[15:0]=400 @bits63:48
        g1[2] = (int)(((unsigned)OUTW & 0xFFFFu) << 16);       // tensor_dim1[15:0] @bits95:80
        g1[3] = (int)(((unsigned)DD2 << 16) | ((unsigned)OUTW >> 16)); // tile_dim0=400 | dim1[31:16]
        g1[4] = 16;                                            // tile_dim1=16 rows, tile_dim2=0
        g1[5] = DD2;                                           // tensor_dim0_stride=400
        g1[6] = 0;
        g1[7] = 0;
        v4i gz4 = {0, 0, 0, 0};
        v8i gz8 = {0, 0, 0, 0, 0, 0, 0, 0};
        __builtin_amdgcn_tensor_load_to_lds(g0, g1, gz4, gz4, gz8, 0);
        __builtin_amdgcn_s_wait_tensorcnt(0);
    }
    __syncthreads();

    const int ng = nt * 16 + lcol;        // global output column for this lane
    const int mt0 = wv * 2;               // this wave's first M tile
    const int m0 = mt0 * 16 + lcol;       // A row, tile 0
    const int m1 = m0 + 16;               // A row, tile 1

    v8f acc0, acc1;
    {
        float bias = db2[ng];
        #pragma unroll
        for (int r = 0; r < 8; ++r) { acc0[r] = bias; acc1[r] = bias; }
    }

    for (int kb = 0; kb < DD2 / 4; ++kb) {
        int k = kb * 4;
        v2f bf, af0, af1;
        bf.x  = sB[lcol][k + koff];
        bf.y  = sB[lcol][k + koff + 1];
        af0.x = a1[(size_t)m0 * DD2 + k + koff];
        af0.y = a1[(size_t)m0 * DD2 + k + koff + 1];
        af1.x = a1[(size_t)m1 * DD2 + k + koff];
        af1.y = a1[(size_t)m1 * DD2 + k + koff + 1];
        acc0 = __builtin_amdgcn_wmma_f32_16x16x4_f32(false, af0, false, bf, (short)0, acc0, false, false);
        acc1 = __builtin_amdgcn_wmma_f32_16x16x4_f32(false, af1, false, bf, (short)0, acc1, false, false);
    }

    #pragma unroll
    for (int r = 0; r < 8; ++r) {
        int mm = mt0 * 16 + r + hi * 8;
        out[(size_t)mm * OUTW + ng]        = acc0[r];
        out[(size_t)(mm + 16) * OUTW + ng] = acc1[r];
    }
}

// ---------------------------------------------------------------------------
extern "C" void kernel_launch(void* const* d_in, const int* in_sizes, int n_in,
                              void* d_out, int out_size, void* d_ws, size_t ws_size,
                              hipStream_t stream) {
    const float* x    = (const float*)d_in[0];
    const float* wih0 = (const float*)d_in[1];
    const float* whh0 = (const float*)d_in[2];
    const float* b0   = (const float*)d_in[3];
    const float* wih1 = (const float*)d_in[4];
    const float* whh1 = (const float*)d_in[5];
    const float* b1   = (const float*)d_in[6];
    const float* dw0  = (const float*)d_in[7];
    const float* db0  = (const float*)d_in[8];
    const float* dw1  = (const float*)d_in[9];
    const float* db1  = (const float*)d_in[10];
    const float* dw2  = (const float*)d_in[11];
    const float* db2  = (const float*)d_in[12];
    float* out = (float*)d_out;

    // workspace layout (floats): h1 final | a0 | a1
    float* h1 = (float*)d_ws;             // 256*64
    float* a0 = h1 + BB * HH;             // 256*350
    float* a1 = a0 + BB * DD1;            // 256*400

    lstm_scan_kernel<<<BB / 16, 256, 0, stream>>>(x, wih0, whh0, b0, wih1, whh1, b1, h1);
    dec0_kernel<<<(BB * DD1 + 255) / 256, 256, 0, stream>>>(h1, dw0, db0, a0);
    dec1_kernel<<<(BB * DD2 + 255) / 256, 256, 0, stream>>>(a0, dw1, db1, a1);
    dec2_kernel<<<OUTW / 16, 256, 0, stream>>>(a1, dw2, db2, out);
}
